// GlobalAttention_75110388072733
// MI455X (gfx1250) — matrix operand order repaired and checked by reading, not compiled, not run on hardware
//
#include <hip/hip_runtime.h>

// ---------------------------------------------------------------------------
// Types for CDNA5 WMMA (wave32, v_wmma_f32_16x16x32_bf16)
// ---------------------------------------------------------------------------
typedef __bf16 bf16_t;
typedef __attribute__((ext_vector_type(8)))  bf16_t v8bf;
typedef __attribute__((ext_vector_type(16))) bf16_t v16bf;
typedef __attribute__((ext_vector_type(8)))  float  v8f;

#define N_GRAPHS 32
#define SEG_LEN  512
#define NUM_HEADS 16
#define DMODEL   1024
#define DHEAD    64
#define NTOK     (N_GRAPHS * SEG_LEN)   // 16384

__device__ __forceinline__ v16bf combine8(v8bf lo, v8bf hi) {
  v16bf r;
#pragma unroll
  for (int i = 0; i < 8; ++i) { r[i] = lo[i]; r[i + 8] = hi[i]; }
  return r;
}

__device__ __forceinline__ v8f wmma_bf16(v16bf a, v16bf b, v8f c) {
  return __builtin_amdgcn_wmma_f32_16x16x32_bf16(
      /*neg_a=*/false, a, /*neg_b=*/false, b,
      /*c_mod=*/(short)0, c, /*reuse_a=*/false, /*reuse_b=*/false);
}

// ---------------------------------------------------------------------------
// CDNA5 async global->LDS copy (ASYNCcnt-tracked, no VGPR round trip).
// Per lane: 16 bytes from `gptr` into LDS byte offset `lds_off`.
// (CDNA5 ISA; VGLOBAL encoding, GV addressing mode)
// ---------------------------------------------------------------------------
__device__ __forceinline__ void async_ld_b128(unsigned lds_off, const void* gptr) {
  asm volatile("global_load_async_to_lds_b128 %0, %1, off"
               :: "v"(lds_off), "v"((unsigned long long)(size_t)gptr)
               : "memory");
}
__device__ __forceinline__ void wait_async0() {
  asm volatile("s_wait_asynccnt 0x0" ::: "memory");
}

// ---------------------------------------------------------------------------
// Kernel 0: f32 -> bf16 conversion (bandwidth-bound, trivial)
// ---------------------------------------------------------------------------
__global__ __launch_bounds__(256)
void cvt_f32_to_bf16(const float* __restrict__ in, bf16_t* __restrict__ out, int n) {
  int i = blockIdx.x * blockDim.x + threadIdx.x;
  int stride = gridDim.x * blockDim.x;
  for (; i < n; i += stride) out[i] = (bf16_t)in[i];
}

// ---------------------------------------------------------------------------
// Kernel 1/3: NT GEMM  C[m,n] = sum_k A[m,k] * B[n,k] + bias[n]
// Block tile 128(M) x 128(N), Ktile = 32, double-buffered LDS fed by
// global_load_async_to_lds_b128. 8 waves in a 4(M) x 2(N) grid; each wave
// computes a 32x64 tile = 8 WMMAs per k-step.
// mode 0: store f32 row-major (out-proj).
// mode 1: scatter bf16 into Q [g,h,s,dh], K [g,h,s,dh], V^T [g,h,dh,s].
// ---------------------------------------------------------------------------
__global__ __launch_bounds__(256)
void gemm_nt_bf16_wmma(const bf16_t* __restrict__ A, const bf16_t* __restrict__ B,
                       const float* __restrict__ bias,
                       int M, int N, int K, int mode,
                       float* __restrict__ outF,
                       bf16_t* __restrict__ qb, bf16_t* __restrict__ kb,
                       bf16_t* __restrict__ vtb) {
  __shared__ __align__(128) bf16_t Asl[2][128 * 32];  // [buf][row*32 + k]
  __shared__ __align__(128) bf16_t Bsl[2][128 * 32];

  const int t    = threadIdx.x;
  const int lane = t & 31;
  const int wave = t >> 5;
  const int lo16 = lane & 15;
  const int hi   = lane >> 4;
  const int wm   = (wave & 3) * 32;   // wave M offset inside block tile
  const int wn   = (wave >> 2) * 64;  // wave N offset inside block tile
  const int mblock = blockIdx.y * 128;
  const int nblock = blockIdx.x * 128;

  // Stage one 128x32 bf16 tile pair: 512 16B chunks each -> 2+2 asyncs/thread.
  auto stage = [&](int buf, int k0) {
    unsigned abase = (unsigned)(size_t)(&Asl[buf][0]);
    unsigned bbase = (unsigned)(size_t)(&Bsl[buf][0]);
#pragma unroll
    for (int i = 0; i < 2; ++i) {
      int c = t + i * 256;            // chunk id; 4 chunks per 32-half row
      int row = c >> 2, col = (c & 3) * 8;
      async_ld_b128(abase + c * 16, A + (size_t)(mblock + row) * K + k0 + col);
      async_ld_b128(bbase + c * 16, B + (size_t)(nblock + row) * K + k0 + col);
    }
  };

  v8f acc[2][4];
#pragma unroll
  for (int i2 = 0; i2 < 2; ++i2)
#pragma unroll
    for (int j = 0; j < 4; ++j)
#pragma unroll
      for (int i = 0; i < 8; ++i) acc[i2][j][i] = 0.f;

  stage(0, 0);
  wait_async0();
  __syncthreads();

  int buf = 0;
  for (int k0 = 0; k0 < K; k0 += 32, buf ^= 1) {
    if (k0 + 32 < K) stage(buf ^ 1, k0 + 32);  // prefetch next tile

    const bf16_t* As = &Asl[buf][0];
    const bf16_t* Bs = &Bsl[buf][0];

    // A fragments (16x32): lane holds row lane&15; K = hi*16..+15 contiguous
    v16bf af[2];
#pragma unroll
    for (int i2 = 0; i2 < 2; ++i2) {
      int arow = wm + i2 * 16 + lo16;
      v8bf alo = *(const v8bf*)(As + arow * 32 + hi * 16);
      v8bf ahi = *(const v8bf*)(As + arow * 32 + hi * 16 + 8);
      af[i2] = combine8(alo, ahi);
    }
#pragma unroll
    for (int j = 0; j < 4; ++j) {
      // B fragment (32x16): lane holds col lane&15, K = hi*16..+15 contiguous.
      int brow = wn + j * 16 + lo16;
      v16bf bfrag = *(const v16bf*)(Bs + brow * 32 + hi * 16);
#pragma unroll
      for (int i2 = 0; i2 < 2; ++i2)
        acc[i2][j] = wmma_bf16(af[i2], bfrag, acc[i2][j]);
    }

    wait_async0();     // next tile landed in LDS
    __syncthreads();   // everyone done reading current buf
  }

  // Epilogue. C/D layout: VGPR i <-> row i + 8*hi; col = lane&15.
#pragma unroll
  for (int i2 = 0; i2 < 2; ++i2) {
#pragma unroll
    for (int j = 0; j < 4; ++j) {
      int n = nblock + wn + j * 16 + lo16;
      float bn = bias ? bias[n] : 0.f;
#pragma unroll
      for (int i = 0; i < 8; ++i) {
        int row = mblock + wm + i2 * 16 + hi * 8 + i;
        float v = acc[i2][j][i] + bn;
        if (mode == 0) {
          outF[(size_t)row * N + n] = v;
        } else {
          int c = n >> 10, h = (n >> 6) & 15, di = n & 63;
          int g = row >> 9, s = row & 511;
          size_t gh = (size_t)g * NUM_HEADS + h;
          bf16_t bv = (bf16_t)v;
          if (c == 0)      qb[(gh * SEG_LEN + s) * DHEAD + di] = bv;
          else if (c == 1) kb[(gh * SEG_LEN + s) * DHEAD + di] = bv;
          else             vtb[(gh * DHEAD + di) * SEG_LEN + s] = bv;  // V^T
        }
      }
    }
  }
}

// ---------------------------------------------------------------------------
// Kernel 2: flash-style block-diagonal attention.
// Grid (S/128, H, G); block = 256 (8 waves); each wave owns 16 query rows.
// K / V^T chunks double-buffered in LDS via async loads; online softmax stats
// per-lane f32 (row r = i + 8*hi lives in a fixed 16-lane half -> shfl_xor
// masks 1,2,4,8 reduce over that row's 16 column-lanes).
// ---------------------------------------------------------------------------
__global__ __launch_bounds__(256)
void attn_flash_wmma(const bf16_t* __restrict__ qg, const bf16_t* __restrict__ kg,
                     const bf16_t* __restrict__ vtg, bf16_t* __restrict__ ctx) {
  __shared__ __align__(128) bf16_t Kt[2][32 * 64];    // [buf][s_local][d]
  __shared__ __align__(128) bf16_t Vt[2][64 * 32];    // [buf][d][s_local]
  __shared__ __align__(128) bf16_t Ps[8][16 * 32];    // per-wave P staging

  const int t    = threadIdx.x;
  const int lane = t & 31;
  const int wave = t >> 5;
  const int lo16 = lane & 15;
  const int hi   = lane >> 4;
  const int g  = blockIdx.z;
  const int h  = blockIdx.y;
  const int qt = blockIdx.x;
  const size_t gh = (size_t)g * NUM_HEADS + h;
  const int qrow0 = qt * 128 + wave * 16;

  // Stage a 32-key chunk: K (32x64) + V^T (64x32) = 256 chunks each -> 2/thread.
  auto stage = [&](int buf, int s0) {
    unsigned kbase = (unsigned)(size_t)(&Kt[buf][0]);
    unsigned vbase = (unsigned)(size_t)(&Vt[buf][0]);
    int krow = t >> 2, kcol = (t & 3) * 8;   // 4 chunks per 64-half K row... 
    // K rows are 64 halves = 128B = 8 chunks; map linearly instead:
    (void)krow; (void)kcol;
    int row8 = t >> 3, col8 = (t & 7) * 8;   // K: 8 chunks per row
    async_ld_b128(kbase + t * 16, kg + (gh * SEG_LEN + s0 + row8) * DHEAD + col8);
    int vrow = t >> 2, vcol = (t & 3) * 8;   // V^T: 4 chunks per 32-half row
    async_ld_b128(vbase + t * 16, vtg + (gh * DHEAD + vrow) * SEG_LEN + s0 + vcol);
  };

  // Q fragments for this wave's 16 rows (dh=64 -> two K=32 fragments).
  const bf16_t* qptr = qg + (gh * SEG_LEN + qrow0 + lo16) * DHEAD;
  v16bf qf[2];
#pragma unroll
  for (int kk = 0; kk < 2; ++kk) {
    v8bf lo = *(const v8bf*)(qptr + kk * 32 + hi * 16);
    v8bf hh = *(const v8bf*)(qptr + kk * 32 + hi * 16 + 8);
    qf[kk] = combine8(lo, hh);
  }

  v8f accv[4];
#pragma unroll
  for (int j = 0; j < 4; ++j)
#pragma unroll
    for (int i = 0; i < 8; ++i) accv[j][i] = 0.f;

  float mrun[8], lrun[8];
#pragma unroll
  for (int i = 0; i < 8; ++i) { mrun[i] = -1e30f; lrun[i] = 0.f; }
  const float scale = 0.125f;  // 1/sqrt(64)

  stage(0, 0);
  wait_async0();
  __syncthreads();

  int buf = 0;
  for (int s0 = 0; s0 < SEG_LEN; s0 += 32, buf ^= 1) {
    if (s0 + 32 < SEG_LEN) stage(buf ^ 1, s0 + 32);

    const bf16_t* Ks = &Kt[buf][0];
    const bf16_t* Vs = &Vt[buf][0];

    // Scores: two 16x16 tiles, each 2 WMMA k-steps over dh=64.
    v8f st[2];
#pragma unroll
    for (int j = 0; j < 2; ++j) {
      v8f s;
#pragma unroll
      for (int i = 0; i < 8; ++i) s[i] = 0.f;
#pragma unroll
      for (int kk = 0; kk < 2; ++kk) {
        v16bf bf = *(const v16bf*)(Ks + (j * 16 + lo16) * 64 + kk * 32 + hi * 16);
        s = wmma_bf16(qf[kk], bf, s);
      }
      st[j] = s;
    }

    // Online softmax update (f32 VALU + shfl reductions over 16-lane halves).
    float p0[8], p1[8], alpha[8];
#pragma unroll
    for (int i = 0; i < 8; ++i) {
      float a0 = st[0][i] * scale;
      float a1 = st[1][i] * scale;
      float mx = fmaxf(a0, a1);
      mx = fmaxf(mx, __shfl_xor(mx, 1, 32));
      mx = fmaxf(mx, __shfl_xor(mx, 2, 32));
      mx = fmaxf(mx, __shfl_xor(mx, 4, 32));
      mx = fmaxf(mx, __shfl_xor(mx, 8, 32));
      float mn = fmaxf(mrun[i], mx);
      float a  = __expf(mrun[i] - mn);
      p0[i] = __expf(a0 - mn);
      p1[i] = __expf(a1 - mn);
      float r = p0[i] + p1[i];
      r += __shfl_xor(r, 1, 32);
      r += __shfl_xor(r, 2, 32);
      r += __shfl_xor(r, 4, 32);
      r += __shfl_xor(r, 8, 32);
      lrun[i] = lrun[i] * a + r;
      mrun[i] = mn;
      alpha[i] = a;
    }
#pragma unroll
    for (int j = 0; j < 4; ++j)
#pragma unroll
      for (int i = 0; i < 8; ++i) accv[j][i] *= alpha[i];

    // C-layout P -> per-wave LDS slab -> reload in A-fragment layout.
    // Wave-local DS traffic is in-order; no barrier needed for own slab.
    bf16_t* pp = Ps[wave];
#pragma unroll
    for (int i = 0; i < 8; ++i) {
      int prow = hi * 8 + i;
      pp[prow * 32 + lo16]      = (bf16_t)p0[i];
      pp[prow * 32 + 16 + lo16] = (bf16_t)p1[i];
    }
    v8bf plo = *(const v8bf*)(pp + lo16 * 32 + hi * 16);
    v8bf phi = *(const v8bf*)(pp + lo16 * 32 + hi * 16 + 8);
    v16bf pf = combine8(plo, phi);

    // ctx += P(16x32) @ V(32x64): 4 WMMAs (B-frags contiguous thanks to V^T).
#pragma unroll
    for (int j = 0; j < 4; ++j) {
      v16bf vf = *(const v16bf*)(Vs + (j * 16 + lo16) * 32 + hi * 16);
      accv[j] = wmma_bf16(pf, vf, accv[j]);
    }

    wait_async0();
    __syncthreads();
  }

  // Normalize and store ctx as bf16 row-major (NTOK x DMODEL).
#pragma unroll
  for (int j = 0; j < 4; ++j) {
    int d = h * DHEAD + j * 16 + lo16;
#pragma unroll
    for (int i = 0; i < 8; ++i) {
      int row = g * SEG_LEN + qrow0 + hi * 8 + i;
      ctx[(size_t)row * DMODEL + d] = (bf16_t)(accv[j][i] / lrun[i]);
    }
  }
}

// ---------------------------------------------------------------------------
// Kernel 4: residual + LayerNorm (one row per block).
// ---------------------------------------------------------------------------
__global__ __launch_bounds__(256)
void resid_layernorm(const float* __restrict__ x, const float* __restrict__ pj,
                     const float* __restrict__ gamma, const float* __restrict__ beta,
                     float* __restrict__ out) {
  __shared__ float ssum[256];
  __shared__ float ssq[256];
  const int row = blockIdx.x;
  const int t = threadIdx.x;
  const float* xr = x + (size_t)row * DMODEL;
  const float* pr = pj + (size_t)row * DMODEL;

  float hv[4];
  float s = 0.f, q = 0.f;
#pragma unroll
  for (int i = 0; i < 4; ++i) {
    int c = t + i * 256;
    float v = xr[c] + pr[c];
    hv[i] = v;
    s += v;
    q += v * v;
  }
  ssum[t] = s;
  ssq[t] = q;
  __syncthreads();
  for (int off = 128; off > 0; off >>= 1) {
    if (t < off) { ssum[t] += ssum[t + off]; ssq[t] += ssq[t + off]; }
    __syncthreads();
  }
  float mean = ssum[0] * (1.f / DMODEL);
  float var  = ssq[0] * (1.f / DMODEL) - mean * mean;
  float inv  = rsqrtf(var + 1e-5f);
  float* orow = out + (size_t)row * DMODEL;
#pragma unroll
  for (int i = 0; i < 4; ++i) {
    int c = t + i * 256;
    orow[c] = (hv[i] - mean) * inv * gamma[c] + beta[c];
  }
}

// ---------------------------------------------------------------------------
// Host-side launch
// ---------------------------------------------------------------------------
extern "C" void kernel_launch(void* const* d_in, const int* in_sizes, int n_in,
                              void* d_out, int out_size, void* d_ws, size_t ws_size,
                              hipStream_t stream) {
  const float* x     = (const float*)d_in[0];
  // d_in[1] = batch (segment ids) — structure is static, not needed.
  const float* w_in  = (const float*)d_in[2];
  const float* b_in  = (const float*)d_in[3];
  const float* w_out = (const float*)d_in[4];
  const float* b_out = (const float*)d_in[5];
  const float* gamma = (const float*)d_in[6];
  const float* beta  = (const float*)d_in[7];
  float* out = (float*)d_out;

  // Workspace carve (bf16 buffers; proj f32 aliases the dead Q/K region).
  bf16_t* w_in_b  = (bf16_t*)d_ws;                         // 3072*1024
  bf16_t* w_out_b = w_in_b  + (size_t)3 * DMODEL * DMODEL; // 1024*1024
  bf16_t* x_b     = w_out_b + (size_t)DMODEL * DMODEL;     // NTOK*1024
  bf16_t* ctx_b   = x_b     + (size_t)NTOK * DMODEL;       // NTOK*1024
  bf16_t* q_b     = ctx_b   + (size_t)NTOK * DMODEL;       // NTOK*1024
  bf16_t* k_b     = q_b     + (size_t)NTOK * DMODEL;
  bf16_t* vt_b    = k_b     + (size_t)NTOK * DMODEL;
  float*  proj    = (float*)q_b;  // 64 MB, reuses Q+K after attention

  // 0) f32 -> bf16 packs
  cvt_f32_to_bf16<<<8192, 256, 0, stream>>>(x, x_b, NTOK * DMODEL);
  cvt_f32_to_bf16<<<2048, 256, 0, stream>>>(w_in, w_in_b, 3 * DMODEL * DMODEL);
  cvt_f32_to_bf16<<<1024, 256, 0, stream>>>(w_out, w_out_b, DMODEL * DMODEL);

  // 1) QKV = x @ w_in^T + b_in, scattered to Q/K/V^T (bf16)
  gemm_nt_bf16_wmma<<<dim3(3 * DMODEL / 128, NTOK / 128), 256, 0, stream>>>(
      x_b, w_in_b, b_in, NTOK, 3 * DMODEL, DMODEL, /*mode=*/1,
      nullptr, q_b, k_b, vt_b);

  // 2) Flash attention per (graph, head, 128-row q tile)
  attn_flash_wmma<<<dim3(SEG_LEN / 128, NUM_HEADS, N_GRAPHS), 256, 0, stream>>>(
      q_b, k_b, vt_b, ctx_b);

  // 3) proj = ctx @ w_out^T + b_out (f32)
  gemm_nt_bf16_wmma<<<dim3(DMODEL / 128, NTOK / 128), 256, 0, stream>>>(
      ctx_b, w_out_b, b_out, NTOK, DMODEL, DMODEL, /*mode=*/0,
      proj, nullptr, nullptr, nullptr);

  // 4) out = LayerNorm(x + proj) * gamma + beta
  resid_layernorm<<<NTOK, 256, 0, stream>>>(x, proj, gamma, beta, out);
}
